// Decoder_48395691491396
// MI455X (gfx1250) — matrix operand order repaired and checked by reading, not compiled, not run on hardware
//
#include <hip/hip_runtime.h>
#include <stddef.h>

// ---------------------------------------------------------------------------
// Social-LSTM decoder, fused per-scene persistent kernel for MI455X (gfx1250).
//
// Roofline: ~420 GFLOP per forward vs ~10 MB of real I/O => strongly compute
// bound. All GEMMs run on v_wmma_f32_16x16x32_bf16 (f32 accumulate). One
// workgroup per scene (96 blocks x 512 threads = 16 wave32s); pooling only
// mixes peds within a scene, so the whole 12-step recurrence runs in a single
// kernel with only block barriers. fp32 weights are converted to bf16 images
// in d_ws once per launch (~4 MB, resident in the 192 MB L2).
//
// Register blocking: 2x2 output tiles for the pool MLPs (~88% of FLOPs) and
// 1x2 for the M=16 GEMMs, cutting WMMA operand traffic to 2 (resp. 3)
// 128-bit loads per WMMA so the matrix pipe, not LDS/VMEM issue, is the limit.
//
// WMMA operand layouts (CDNA5 ISA 7.12.2), all served by contiguous 16B loads:
//   A (16x32 bf16): lane row m=lane%16, chunks [k0+half*8,+8) and +16.
//   B (32x16 bf16): lane col n=lane%16, chunks [k0+half*8,+8) and +16 of W[n][k],
//     i.e. exactly the PyTorch (out,in) weight layout -- no transpose needed.
//   C/D (16x16 f32): m = r + 8*half, n = lane%16.
// ---------------------------------------------------------------------------

#define OBS    8
#define SEQN   12
#define NSCENE 96
#define NPED   16
#define NB     (NSCENE * NPED)   // 1536
#define HD     128
#define ED     64
#define BOTD   1024
#define MIDD   512

#define NTHREADS 512
#define NWAVES   (NTHREADS / 32)

typedef __bf16 bf16_t;
typedef __attribute__((ext_vector_type(16))) __bf16        v16bf;
typedef __attribute__((ext_vector_type(8)))  float         v8f;
typedef __attribute__((ext_vector_type(4)))  unsigned int  u32x4;

union V16BF { v16bf v; u32x4 q[2]; };

#define WMMA_BF16(A, Bv, C) \
  __builtin_amdgcn_wmma_f32_16x16x32_bf16(false, (A), false, (Bv), (short)0, (C), false, false)

// fp32 -> bf16, round-to-nearest-even
__device__ __forceinline__ bf16_t f2bf(float f) {
  union { float f; unsigned u; } x; x.f = f;
  unsigned r = x.u + 0x7fffu + ((x.u >> 16) & 1u);
  unsigned short h = (unsigned short)(r >> 16);
  bf16_t b; __builtin_memcpy(&b, &h, 2); return b;
}

// A-operand fetch from row-major bf16 (LDS): two 16-byte chunks per lane.
__device__ __forceinline__ void ldA(V16BF& a, const bf16_t* base, int stride,
                                    int m, int k0, int half) {
  const bf16_t* p = base + m * stride + k0 + half * 8;
  a.q[0] = *(const u32x4*)(p);        // k0+half*8   .. +7   (VGPR 0-3)
  a.q[1] = *(const u32x4*)(p + 16);   // k0+16+half*8.. +7   (VGPR 4-7)
}

__device__ __forceinline__ void ldB(V16BF& b, const bf16_t* wrow_k) {
  b.q[0] = *(const u32x4*)(wrow_k);
  b.q[1] = *(const u32x4*)(wrow_k + 16);
}

// --- M=16 GEMM, 1x2 register blocking: A shared across two N tiles. --------
template <typename FA, typename FD>
__device__ __forceinline__ void wmma_gemm_1x2(int Nt, int Kt,
                                              const bf16_t* __restrict__ W, int ldw,
                                              FA&& loadA, FD&& emitD) {
  const int lane = threadIdx.x & 31;
  const int wid  = threadIdx.x >> 5;
  const int nlo  = lane & 15;
  const int half = lane >> 4;
  const int nsup = Nt >> 1;
  for (int t = wid; t < nsup; t += NWAVES) {
    const int nt0 = t * 2;
    v8f acc0 = {0.f, 0.f, 0.f, 0.f, 0.f, 0.f, 0.f, 0.f};
    v8f acc1 = acc0;
    const bf16_t* w0 = W + (size_t)(nt0 * 16 + nlo) * ldw + half * 8;
    const bf16_t* w1 = w0 + (size_t)16 * ldw;
    for (int kt = 0; kt < Kt; ++kt) {
      V16BF a, b0, b1;
      loadA(a, nlo, kt * 32, half);
      ldB(b0, w0 + kt * 32);
      ldB(b1, w1 + kt * 32);
      acc0 = WMMA_BF16(a.v, b0.v, acc0);
      acc1 = WMMA_BF16(a.v, b1.v, acc1);
    }
    emitD(acc0, nt0,     nlo, half);
    emitD(acc1, nt0 + 1, nlo, half);
  }
}

// --- Large GEMM, 2x2 register blocking: 4 WMMAs per 8 operand loads. -------
template <typename FA, typename FD>
__device__ __forceinline__ void wmma_gemm_2x2(int Mt, int Nt, int Kt,
                                              const bf16_t* __restrict__ W, int ldw,
                                              FA&& loadA, FD&& emitD) {
  const int lane = threadIdx.x & 31;
  const int wid  = threadIdx.x >> 5;
  const int nlo  = lane & 15;
  const int half = lane >> 4;
  const int msup = Mt >> 1, nsup = Nt >> 1;
  const int tiles = msup * nsup;
  for (int t = wid; t < tiles; t += NWAVES) {
    const int ms = t / nsup;
    const int ns = t - ms * nsup;
    const int mt0 = ms * 2, nt0 = ns * 2;
    v8f a00 = {0.f, 0.f, 0.f, 0.f, 0.f, 0.f, 0.f, 0.f};
    v8f a01 = a00, a10 = a00, a11 = a00;
    const bf16_t* w0 = W + (size_t)(nt0 * 16 + nlo) * ldw + half * 8;
    const bf16_t* w1 = w0 + (size_t)16 * ldw;
    for (int kt = 0; kt < Kt; ++kt) {
      V16BF av0, av1, b0, b1;
      loadA(av0, mt0 * 16 + nlo,      kt * 32, half);
      loadA(av1, mt0 * 16 + 16 + nlo, kt * 32, half);
      ldB(b0, w0 + kt * 32);
      ldB(b1, w1 + kt * 32);
      a00 = WMMA_BF16(av0.v, b0.v, a00);
      a01 = WMMA_BF16(av0.v, b1.v, a01);
      a10 = WMMA_BF16(av1.v, b0.v, a10);
      a11 = WMMA_BF16(av1.v, b1.v, a11);
    }
    emitD(a00, mt0,     nt0,     nlo, half);
    emitD(a01, mt0,     nt0 + 1, nlo, half);
    emitD(a10, mt0 + 1, nt0,     nlo, half);
    emitD(a11, mt0 + 1, nt0 + 1, nlo, half);
  }
}

struct KParams {
  const float *obs_traj, *obs_rel, *h0, *c0;
  const float *W_emb, *b_emb, *b_ih, *b_hh, *W_pos, *b_pos;
  const float *Wp_emb, *bp_emb, *bp1, *bp2, *bm1, *bm2;
  const bf16_t *Wih, *Whh, *Wp1, *Wp2, *Wm1, *Wm2;
  float *out;
};

// Dynamic LDS layout (bytes). region1 reused: gates(f32)/out1(bf16)/mid(bf16).
// region2 reused: emb(bf16)/pooled(bf16).
#define R1_BYTES   (128 * 512 * 2)                 // 131072
#define R2_BYTES   (256 * 64 * 2)                  // 32768
#define SMEM_BYTES (R1_BYTES + R2_BYTES + NPED*HD*4 + NPED*HD*4 + \
                    NPED*HD*2 + NPED*ED*2 + 128 + 128)   // 186624 <= 320K/WGP

__global__ void __launch_bounds__(NTHREADS, 1) cvt_f32_bf16_kernel(
    const float* __restrict__ s, bf16_t* __restrict__ d, int n) {
  int i = blockIdx.x * blockDim.x + threadIdx.x;
  if (i < n) d[i] = f2bf(s[i]);
}

__global__ void __launch_bounds__(NTHREADS, 1) decoder_scene_kernel(KParams p) {
  extern __shared__ __align__(16) char smem[];
  char* sp = smem;
  char*   region1   = sp;            sp += R1_BYTES;
  bf16_t* emb_bf    = (bf16_t*)sp;   sp += R2_BYTES;
  float*  h_f       = (float*)sp;    sp += NPED * HD * 4;
  float*  c_f       = (float*)sp;    sp += NPED * HD * 4;
  bf16_t* h_bf      = (bf16_t*)sp;   sp += NPED * HD * 2;
  bf16_t* dec_bf    = (bf16_t*)sp;   sp += NPED * ED * 2;
  float*  pos       = (float*)sp;    sp += 128;
  float*  relpos    = (float*)sp;    sp += 128;
  float*  gatesf    = (float*)region1;     // 16x512 f32
  bf16_t* out1      = (bf16_t*)region1;    // 128x512 bf16 (half of pool rows)
  bf16_t* mid       = (bf16_t*)region1;    // 16x1024 bf16
  bf16_t* pooled_bf = emb_bf;              // 16x1024 bf16 (emb dead by then)

  const int tid   = threadIdx.x;
  const int gbase = blockIdx.x * NPED;     // first ped of this scene

  // ---- init: h, c, last_pos, dec_in0 = emb(obs_traj_rel[-1]) ----
  for (int idx = tid; idx < NPED * HD; idx += NTHREADS) {
    int ped = idx >> 7, u = idx & 127;
    float hv = p.h0[(size_t)(gbase + ped) * HD + u];
    float cv = p.c0[(size_t)(gbase + ped) * HD + u];
    h_f[idx] = hv; h_bf[idx] = f2bf(hv); c_f[idx] = cv;
  }
  if (tid < 32)
    pos[tid] = p.obs_traj[((size_t)(OBS - 1) * NB + gbase + (tid >> 1)) * 2 + (tid & 1)];
  for (int idx = tid; idx < NPED * ED; idx += NTHREADS) {
    int ped = idx >> 6, e = idx & 63;
    float r0 = p.obs_rel[((size_t)(OBS - 1) * NB + gbase + ped) * 2 + 0];
    float r1 = p.obs_rel[((size_t)(OBS - 1) * NB + gbase + ped) * 2 + 1];
    dec_bf[idx] = f2bf(p.b_emb[e] + r0 * p.W_emb[e * 2] + r1 * p.W_emb[e * 2 + 1]);
  }
  __syncthreads();

  for (int step = 0; step < SEQN; ++step) {
    // ---- 1. LSTM gates: (16x64 @ 64x512) + (16x128 @ 128x512) ----
    wmma_gemm_1x2(32, 2, p.Wih, ED,
      [&](V16BF& a, int m, int k0, int half) { ldA(a, dec_bf, ED, m, k0, half); },
      [&](v8f acc, int nt, int nlo, int half) {
        int n = nt * 16 + nlo; float bb = p.b_ih[n] + p.b_hh[n];
        #pragma unroll
        for (int r = 0; r < 8; ++r) gatesf[(r + 8 * half) * 512 + n] = acc[r] + bb;
      });
    wmma_gemm_1x2(32, 4, p.Whh, HD,            // same supertile->wave map: no race
      [&](V16BF& a, int m, int k0, int half) { ldA(a, h_bf, HD, m, k0, half); },
      [&](v8f acc, int nt, int nlo, int half) {
        int n = nt * 16 + nlo;
        #pragma unroll
        for (int r = 0; r < 8; ++r) gatesf[(r + 8 * half) * 512 + n] += acc[r];
      });
    __syncthreads();

    // ---- 2. LSTM pointwise (i,f,g,o), update c, h ----
    for (int idx = tid; idx < NPED * HD; idx += NTHREADS) {
      int ped = idx >> 7, u = idx & 127;
      const float* gr = gatesf + ped * 512;
      float si = 1.f / (1.f + expf(-gr[u]));
      float sf = 1.f / (1.f + expf(-gr[128 + u]));
      float tg = tanhf(gr[256 + u]);
      float so = 1.f / (1.f + expf(-gr[384 + u]));
      float c = sf * c_f[idx] + si * tg;
      float h = so * tanhf(c);
      c_f[idx] = c; h_f[idx] = h; h_bf[idx] = f2bf(h);
    }
    __syncthreads();

    // ---- 3. rel_pos = h @ W_pos^T + b_pos ; curr_pos ; emit output ----
    if (tid < 32) {
      int ped = tid >> 1, d = tid & 1;
      float acc = p.b_pos[d];
      const float* wr = p.W_pos + d * HD;
      const float* hr = h_f + ped * HD;
      for (int k = 0; k < HD; ++k) acc += hr[k] * wr[k];
      relpos[tid] = acc;
      pos[tid] += acc;                                   // curr_pos
      p.out[((size_t)step * NB + gbase + ped) * 2 + d] = acc;
    }
    __syncthreads();

    // ---- 4. next dec_in embedding + pairwise pool embedding ----
    for (int idx = tid; idx < NPED * ED; idx += NTHREADS) {
      int ped = idx >> 6, e = idx & 63;
      dec_bf[idx] = f2bf(p.b_emb[e] + relpos[ped * 2] * p.W_emb[e * 2]
                                    + relpos[ped * 2 + 1] * p.W_emb[e * 2 + 1]);
    }
    for (int idx = tid; idx < 256 * ED; idx += NTHREADS) {
      int row = idx >> 6, e = idx & 63;
      int i = row >> 4, j = row & 15;
      float r0 = pos[j * 2] - pos[i * 2];
      float r1 = pos[j * 2 + 1] - pos[i * 2 + 1];
      emb_bf[idx] = f2bf(p.bp_emb[e] + r0 * p.Wp_emb[e * 2] + r1 * p.Wp_emb[e * 2 + 1]);
    }
    __syncthreads();

    // ---- 5. Pool MLP in two 128-row halves; fused relu+neighbor-max ----
    for (int mo = 0; mo < 256; mo += 128) {
      // MLP1: rows = (i,j) pairs, K=192 pieced from [emb(i,j) | h(j)]
      wmma_gemm_2x2(8, 32, 6, p.Wp1, ED + HD,
        [&](V16BF& a, int m, int k0, int half) {
          int gm = mo + m;
          if (k0 < ED) ldA(a, emb_bf, ED, gm, k0, half);
          else         ldA(a, h_bf, HD, gm & 15, k0 - ED, half);
        },
        [&](v8f acc, int mt, int nt, int nlo, int half) {
          int n = nt * 16 + nlo; float bb = p.bp1[n];
          #pragma unroll
          for (int r = 0; r < 8; ++r) {
            float v = acc[r] + bb; v = v > 0.f ? v : 0.f;
            out1[(mt * 16 + r + 8 * half) * MIDD + n] = f2bf(v);
          }
        });
      __syncthreads();
      // MLP2: each 16-row D-tile holds all 16 neighbors j of one ped i ->
      // relu + max over j fused into the epilogue (8 regs + shfl_xor 16).
      wmma_gemm_2x2(8, 64, 16, p.Wp2, MIDD,
        [&](V16BF& a, int m, int k0, int half) { ldA(a, out1, MIDD, m, k0, half); },
        [&](v8f acc, int mt, int nt, int nlo, int half) {
          int n = nt * 16 + nlo; float bb = p.bp2[n];
          float mx = 0.f;                                // relu folded into max
          #pragma unroll
          for (int r = 0; r < 8; ++r) { float v = acc[r] + bb; mx = v > mx ? v : mx; }
          float o = __shfl_xor(mx, 16, 32);
          mx = mx > o ? mx : o;
          if (half == 0) pooled_bf[(mo / 16 + mt) * BOTD + n] = f2bf(mx);
        });
      __syncthreads();
    }

    // ---- 6. decoder MLP: relu(relu([h|pool] @ Wm1^T) @ Wm2^T) -> new h ----
    wmma_gemm_1x2(64, 36, p.Wm1, HD + BOTD,
      [&](V16BF& a, int m, int k0, int half) {
        if (k0 < HD) ldA(a, h_bf, HD, m, k0, half);
        else         ldA(a, pooled_bf, BOTD, m, k0 - HD, half);
      },
      [&](v8f acc, int nt, int nlo, int half) {
        int n = nt * 16 + nlo; float bb = p.bm1[n];
        #pragma unroll
        for (int r = 0; r < 8; ++r) {
          float v = acc[r] + bb; v = v > 0.f ? v : 0.f;
          mid[(r + 8 * half) * BOTD + n] = f2bf(v);
        }
      });
    __syncthreads();
    wmma_gemm_1x2(8, 32, p.Wm2, BOTD,
      [&](V16BF& a, int m, int k0, int half) { ldA(a, mid, BOTD, m, k0, half); },
      [&](v8f acc, int nt, int nlo, int half) {
        int n = nt * 16 + nlo; float bb = p.bm2[n];
        #pragma unroll
        for (int r = 0; r < 8; ++r) {
          float v = acc[r] + bb; v = v > 0.f ? v : 0.f;
          int row = r + 8 * half;
          h_f[row * HD + n] = v; h_bf[row * HD + n] = f2bf(v);
        }
      });
    __syncthreads();
  }

  // ---- final hidden state (1,B,H) appended after (SEQ,B,2) rels ----
  for (int idx = tid; idx < NPED * HD; idx += NTHREADS)
    p.out[(size_t)SEQN * NB * 2 + (size_t)(gbase + (idx >> 7)) * HD + (idx & 127)] = h_f[idx];
}

extern "C" void kernel_launch(void* const* d_in, const int* in_sizes, int n_in,
                              void* d_out, int out_size, void* d_ws, size_t ws_size,
                              hipStream_t stream) {
  (void)in_sizes; (void)n_in; (void)out_size; (void)ws_size;

  const float* obs_traj = (const float*)d_in[0];
  const float* obs_rel  = (const float*)d_in[1];
  // d_in[2] typeID_seq, d_in[5] seq_start_end: unused by the math
  const float* h0    = (const float*)d_in[3];
  const float* c0    = (const float*)d_in[4];
  const float* W_emb = (const float*)d_in[6];  const float* b_emb = (const float*)d_in[7];
  const float* W_ih  = (const float*)d_in[8];  const float* b_ih  = (const float*)d_in[9];
  const float* W_hh  = (const float*)d_in[10]; const float* b_hh  = (const float*)d_in[11];
  const float* W_pos = (const float*)d_in[12]; const float* b_pos = (const float*)d_in[13];
  const float* Wp_emb= (const float*)d_in[14]; const float* bp_emb= (const float*)d_in[15];
  const float* Wp1   = (const float*)d_in[16]; const float* bp1   = (const float*)d_in[17];
  const float* Wp2   = (const float*)d_in[18]; const float* bp2   = (const float*)d_in[19];
  const float* Wm1   = (const float*)d_in[20]; const float* bm1   = (const float*)d_in[21];
  const float* Wm2   = (const float*)d_in[22]; const float* bm2   = (const float*)d_in[23];

  // bf16 weight images in d_ws (~4 MB total; re-converted every call =>
  // deterministic and graph-capture safe).
  size_t off = 0;
  auto wsa = [&](size_t n) {
    bf16_t* q = (bf16_t*)((char*)d_ws + off);
    off += ((n * sizeof(bf16_t)) + 255) & ~(size_t)255;
    return q;
  };
  bf16_t* Wih_bf = wsa(512 * ED);
  bf16_t* Whh_bf = wsa(512 * HD);
  bf16_t* Wp1_bf = wsa(MIDD * (ED + HD));
  bf16_t* Wp2_bf = wsa(BOTD * MIDD);
  bf16_t* Wm1_bf = wsa(BOTD * (HD + BOTD));
  bf16_t* Wm2_bf = wsa(HD * BOTD);

  auto cvt = [&](const float* s, bf16_t* d, int n) {
    cvt_f32_bf16_kernel<<<(n + NTHREADS - 1) / NTHREADS, NTHREADS, 0, stream>>>(s, d, n);
  };
  cvt(W_ih, Wih_bf, 512 * ED);
  cvt(W_hh, Whh_bf, 512 * HD);
  cvt(Wp1,  Wp1_bf, MIDD * (ED + HD));
  cvt(Wp2,  Wp2_bf, BOTD * MIDD);
  cvt(Wm1,  Wm1_bf, BOTD * (HD + BOTD));
  cvt(Wm2,  Wm2_bf, HD * BOTD);

  KParams kp;
  kp.obs_traj = obs_traj; kp.obs_rel = obs_rel; kp.h0 = h0; kp.c0 = c0;
  kp.W_emb = W_emb; kp.b_emb = b_emb; kp.b_ih = b_ih; kp.b_hh = b_hh;
  kp.W_pos = W_pos; kp.b_pos = b_pos; kp.Wp_emb = Wp_emb; kp.bp_emb = bp_emb;
  kp.bp1 = bp1; kp.bp2 = bp2; kp.bm1 = bm1; kp.bm2 = bm2;
  kp.Wih = Wih_bf; kp.Whh = Whh_bf; kp.Wp1 = Wp1_bf; kp.Wp2 = Wp2_bf;
  kp.Wm1 = Wm1_bf; kp.Wm2 = Wm2_bf;
  kp.out = (float*)d_out;

  (void)hipFuncSetAttribute((const void*)decoder_scene_kernel,
                            hipFuncAttributeMaxDynamicSharedMemorySize,
                            (int)SMEM_BYTES);
  decoder_scene_kernel<<<NSCENE, NTHREADS, SMEM_BYTES, stream>>>(kp);
}
